// TrajectoryGenerator_68590627717129
// MI455X (gfx1250) — matrix operand order repaired and checked by reading, not compiled, not run on hardware
//
#include <hip/hip_runtime.h>
#include <math.h>

#define BDIM  256
#define BATCH 512
#define NAG   64
#define TOBS  8
#define TPRED 12
#define DEMB  16
#define DENC  32
#define DMLP  128

typedef __attribute__((ext_vector_type(16))) _Float16 v16h;
typedef __attribute__((ext_vector_type(8)))  _Float16 v8h;
typedef __attribute__((ext_vector_type(8)))  float    v8f;
typedef __attribute__((ext_vector_type(4)))  float    v4f;

#define ACT_NONE 0
#define ACT_RELU 1
#define ACT_TANH 2

#define WMMA_F16(a, b, c) \
  __builtin_amdgcn_wmma_f32_16x16x32_f16(false, (a), false, (b), (short)0, (c), false, false)

// ---------------------------------------------------------------------------
// Parameter bundle (all global f32 pointers)
// ---------------------------------------------------------------------------
struct P {
  const float *x, *dmat, *bmat, *hmat, *imask, *omask, *mean, *var;
  const float *emb_W, *emb_b;
  const float *encl_Wih, *encl_Whh, *encl_bih, *encl_bhh;
  const float *decl_Wih, *decl_Whh, *decl_bih, *decl_bhh;
  const float *enc2att_W, *enc2att_b, *att2enc_W, *att2enc_b;
  const float *dec2att_W, *dec2att_b, *att2dec_W, *att2dec_b;
  const float *encsp_W, *encsp_b, *decsp_W, *decsp_b;
  const float *tatt_W, *tatt_b;
  const float *out1_W, *out1_b, *out2_W, *out2_b;
  const float *domain;
};

// ---------------------------------------------------------------------------
// WMMA fragment helpers (wave32, 16x16x32 f16 -> f32). Per-lane fragment
// layouts are contiguous identity maps:
//   A (16-bit, lane = M dup): halves h   -> k = k0+16*hi+h (one contiguous run)
//   B (16-bit, lane = N dup): halves h   -> k = k0+16*hi+h (one 32B run)
//   C/D: lane%16 = N, vgpr r holds M = r + 8*(lane/16)
// All K are multiples of 32 (K=16 operands zero-padded at rest) -> no guards.
// ---------------------------------------------------------------------------

// A operand from f32 LDS activations: 4x contiguous v4f loads + cvt
__device__ __forceinline__ v16h fragA(const float* A, int lda, int m0, int k0) {
  int lane = threadIdx.x & 31;
  int m = m0 + (lane & 15);
  int hi = lane >> 4;
  const v4f* s0 = (const v4f*)(A + m * lda + k0 + hi * 16);       // k = 0..15 (+16*hi)
  v4f c0 = s0[0], c1 = s0[1], c2 = s0[2], c3 = s0[3];
  v16h a;
#pragma unroll
  for (int i = 0; i < 4; ++i) {
    a[i]      = (_Float16)c0[i];
    a[4 + i]  = (_Float16)c1[i];
    a[8 + i]  = (_Float16)c2[i];
    a[12 + i] = (_Float16)c3[i];
  }
  return a;
}
// B operand from LDS-staged f16 weights: one contiguous 32B run per lane
__device__ __forceinline__ v16h fragB_H(const _Float16* W, int ldw, int n0, int k0) {
  int lane = threadIdx.x & 31;
  int n = n0 + (lane & 15);
  int hi = lane >> 4;
  const v8h* sp = (const v8h*)(W + n * ldw + k0 + hi * 16);
  v8h lo = sp[0], hh = sp[1];
  return __builtin_shufflevector(lo, hh, 0, 1, 2, 3, 4, 5, 6, 7,
                                 8, 9, 10, 11, 12, 13, 14, 15);
}
// B operand from f32 LDS activation, element (k,n) = Bm[k*ldb + n] (strided)
__device__ __forceinline__ v16h fragB_N(const float* Bm, int ldb, int n0, int k0) {
  int lane = threadIdx.x & 31;
  int n = n0 + (lane & 15);
  int hi = lane >> 4;
  v16h b;
#pragma unroll
  for (int h = 0; h < 16; ++h) b[h] = (_Float16)Bm[(k0 + 16 * hi + h) * ldb + n];
  return b;
}
template <int ACT>
__device__ __forceinline__ void storeD(float* D, int ldd, int m0, int n0, v8f acc,
                                       const float* bias) {
  int lane = threadIdx.x & 31;
  int n = n0 + (lane & 15);
  int hi = lane >> 4;
  float bv = bias ? bias[n] : 0.0f;
#pragma unroll
  for (int r = 0; r < 8; ++r) {
    float v = acc[r] + bv;
    if (ACT == ACT_RELU) v = fmaxf(v, 0.0f);
    else if (ACT == ACT_TANH) v = tanhf(v);
    D[(m0 + r + hi * 8) * ldd + n] = v;
  }
}

// D[M][N] = act(A[M][K] @ W[N][K]^T + bias); 8 waves cooperate on 16x16 tiles
template <int M, int N, int K, int ACT>
__device__ void gemm_w(const float* A, int lda, const _Float16* W, int ldw,
                       const float* bias, float* D, int ldd) {
  constexpr int TM = M >> 4, TN = N >> 4;
  int wave = threadIdx.x >> 5;
  for (int t = wave; t < TM * TN; t += (BDIM >> 5)) {
    int m0 = (t % TM) << 4, n0 = (t / TM) << 4;
    v8f acc = {0.f, 0.f, 0.f, 0.f, 0.f, 0.f, 0.f, 0.f};
#pragma unroll
    for (int k0 = 0; k0 < K; k0 += 32) {
      v16h a = fragA(A, lda, m0, k0);
      v16h b = fragB_H(W, ldw, n0, k0);
      acc = WMMA_F16(a, b, acc);
    }
    storeD<ACT>(D, ldd, m0, n0, acc, bias);
  }
}
// D[M][N] = A[M][K] @ Bm[K][N]   (both LDS f32, k-major B)
template <int M, int N, int K>
__device__ void gemm_a(const float* A, int lda, const float* Bm, int ldb,
                       float* D, int ldd) {
  constexpr int TM = M >> 4, TN = N >> 4;
  int wave = threadIdx.x >> 5;
  for (int t = wave; t < TM * TN; t += (BDIM >> 5)) {
    int m0 = (t % TM) << 4, n0 = (t / TM) << 4;
    v8f acc = {0.f, 0.f, 0.f, 0.f, 0.f, 0.f, 0.f, 0.f};
#pragma unroll
    for (int k0 = 0; k0 < K; k0 += 32) {
      v16h a = fragA(A, lda, m0, k0);
      v16h b = fragB_N(Bm, ldb, n0, k0);
      acc = WMMA_F16(a, b, acc);
    }
    storeD<ACT_NONE>(D, ldd, m0, n0, acc, nullptr);
  }
}
// G[64][128] = XE32[64][32] @ Wih32^T + H[64][32] @ Whh^T  (all K=32, zero-padded)
__device__ void gemm_lstm(const float* XE32, const _Float16* Wih32, const float* H,
                          const _Float16* Whh, float* G) {
  int wave = threadIdx.x >> 5;
  for (int t = wave; t < 32; t += (BDIM >> 5)) {
    int m0 = (t & 3) << 4, n0 = (t >> 2) << 4;
    v8f acc = {0.f, 0.f, 0.f, 0.f, 0.f, 0.f, 0.f, 0.f};
    v16h a0 = fragA(XE32, 32, m0, 0);
    v16h b0 = fragB_H(Wih32, 32, n0, 0);
    acc = WMMA_F16(a0, b0, acc);
    v16h a1 = fragA(H, DENC, m0, 0);
    v16h b1 = fragB_H(Whh, DENC, n0, 0);
    acc = WMMA_F16(a1, b1, acc);
    storeD<ACT_NONE>(G, 4 * DENC, m0, n0, acc, nullptr);
  }
}

__device__ __forceinline__ float sigm(float x) { return 1.0f / (1.0f + expf(-x)); }
__device__ __forceinline__ float mod360(float x) {
  float m = fmodf(x, 360.0f);
  return (m < 0.0f) ? m + 360.0f : m;
}

// Spatial attention weight matrix WM[64][64], normalized rows.
__device__ void spatial_weights(float* WM, float* RS, const float* MK, const float* dom,
                                bool useGlobal, const P& p, int b, int tstep,
                                const float* CUR, const float* PRV, float* HEAD) {
  int tid = threadIdx.x;
  if (!useGlobal) {
    if (tid < NAG) {
      float dx = CUR[tid * 2 + 0] - PRV[tid * 2 + 0];
      float dy = CUR[tid * 2 + 1] - PRV[tid * 2 + 1];
      HEAD[tid] = mod360(57.29577951308232f * atan2f(dy, dx));
    }
    __syncthreads();
  }
  for (int pix = tid; pix < NAG * NAG; pix += BDIM) {
    int i = pix >> 6, j = pix & 63;
    float d, bm, hm;
    if (useGlobal) {
      size_t g = (((size_t)b * NAG + i) * TOBS + tstep) * NAG + j;
      d = p.dmat[g]; bm = p.bmat[g]; hm = p.hmat[g];
    } else {
      float rx = CUR[j * 2 + 0] - CUR[i * 2 + 0];
      float ry = CUR[j * 2 + 1] - CUR[i * 2 + 1];
      d = sqrtf(rx * rx + ry * ry + 1e-12f);
      float ang = mod360(57.29577951308232f * atan2f(ry, rx));
      bm = mod360(ang - HEAD[i]);
      hm = mod360(HEAD[j] - HEAD[i]);
    }
    int hi = (int)floorf(hm * (1.0f / 30.0f)); hi = hi < 0 ? 0 : (hi > 11 ? 11 : hi);
    int bi = (int)floorf(bm * (1.0f / 30.0f)); bi = bi < 0 ? 0 : (bi > 11 ? 11 : bi);
    float wv = dom[hi * 12 + bi] - d;
    wv = wv > 0.0f ? wv : 0.0f;
    wv *= MK[i] * MK[j];
    if (i == j) wv = 0.0f;
    WM[pix] = wv;
  }
  __syncthreads();
  if (tid < NAG) {
    float s = 0.0f;
    for (int j = 0; j < NAG; ++j) s += WM[tid * NAG + j];
    RS[tid] = 1.0f / (s + 1e-14f);
  }
  __syncthreads();
  for (int pix = tid; pix < NAG * NAG; pix += BDIM) WM[pix] *= RS[pix >> 6];
  __syncthreads();
}

// ---------------------------------------------------------------------------
// LDS layout (float offsets)
// ---------------------------------------------------------------------------
#define OFF_H    0        // 2048 : h state [64][32]
#define OFF_C    2048     // 2048 : c state
#define OFF_XE   4096     // 2048 : embedded input [64][32], cols 16..31 zero
#define OFF_G    6144     // 8192 : LSTM gates [64][128] (also MLP hidden)
#define OFF_HA   14336    // 2048 : attention features [64][32]
#define OFF_CAT  16384    // 4096 : concat buffer [64][64]
#define OFF_WM   20480    // 4096 : spatial weights [64][64]
#define OFF_SEQ  24576    // 16384: encoder sequence [64][8][32]
#define OFF_CUR  40960    // 128
#define OFF_PRV  41088    // 128
#define OFF_HEAD 41216    // 64
#define OFF_RS   41280    // 64
#define OFF_MK   41344    // 64
#define OFF_XY   41408    // 128
#define OFF_WF16 41536    // 15360 float-slots = 30720 f16 staged weights
#define SMEM_FLOATS 56896 // ~222 KB < 320 KB/WGP

// staged f16 weight offsets (in halves); Wih padded to [128][32]
#define WO_ENCL_WIH 0       // 4096 (padded K 16->32)
#define WO_ENCL_WHH 4096    // 4096
#define WO_DECL_WIH 8192    // 4096 (padded)
#define WO_DECL_WHH 12288   // 4096
#define WO_ENC2ATT  16384   // 1024
#define WO_ATT2ENC  17408   // 1024
#define WO_DEC2ATT  18432   // 1024
#define WO_ATT2DEC  19456   // 1024
#define WO_ENCSP    20480   // 2048
#define WO_DECSP    22528   // 2048
#define WO_TATT     24576   // 2048
#define WO_OUT1     26624   // 4096

#define PRED_SZ (BATCH * NAG * TPRED * 2)

__device__ __forceinline__ void stage16(_Float16* dst, const float* src, int n) {
  for (int i = threadIdx.x; i < n; i += BDIM) dst[i] = (_Float16)src[i];
}
// stage [rows][kin] -> [rows][32] zero-padded
__device__ __forceinline__ void stage16_pad32(_Float16* dst, const float* src, int rows, int kin) {
  for (int i = threadIdx.x; i < rows * 32; i += BDIM) {
    int n = i >> 5, k = i & 31;
    dst[i] = (k < kin) ? (_Float16)src[n * kin + k] : (_Float16)0.0f;
  }
}

__global__ __launch_bounds__(BDIM, 1)
void traj_forward_kernel(P p, float* __restrict__ out) {
  extern __shared__ float sm[];
  const int b = blockIdx.x;
  const int tid = threadIdx.x;

  float* H   = sm + OFF_H;
  float* C   = sm + OFF_C;
  float* XE  = sm + OFF_XE;   // [64][32], upper half stays zero
  float* G   = sm + OFF_G;
  float* HA  = sm + OFF_HA;
  float* CAT = sm + OFF_CAT;
  float* WM  = sm + OFF_WM;
  float* SEQ = sm + OFF_SEQ;
  float* CUR = sm + OFF_CUR;
  float* PRV = sm + OFF_PRV;
  float* HEAD= sm + OFF_HEAD;
  float* RS  = sm + OFF_RS;
  float* MK  = sm + OFF_MK;
  float* XY  = sm + OFF_XY;
  _Float16* WH = (_Float16*)(sm + OFF_WF16);

  // ---- one-time: stage all GEMM weights into LDS as f16 ----
  stage16_pad32(WH + WO_ENCL_WIH, p.encl_Wih, 4 * DENC, DEMB);
  stage16(WH + WO_ENCL_WHH, p.encl_Whh, 4 * DENC * DENC);
  stage16_pad32(WH + WO_DECL_WIH, p.decl_Wih, 4 * DENC, DEMB);
  stage16(WH + WO_DECL_WHH, p.decl_Whh, 4 * DENC * DENC);
  stage16(WH + WO_ENC2ATT,  p.enc2att_W, 32 * 32);
  stage16(WH + WO_ATT2ENC,  p.att2enc_W, 32 * 32);
  stage16(WH + WO_DEC2ATT,  p.dec2att_W, 32 * 32);
  stage16(WH + WO_ATT2DEC,  p.att2dec_W, 32 * 32);
  stage16(WH + WO_ENCSP,    p.encsp_W, 32 * 64);
  stage16(WH + WO_DECSP,    p.decsp_W, 32 * 64);
  stage16(WH + WO_TATT,     p.tatt_W, 32 * 64);
  stage16(WH + WO_OUT1,     p.out1_W, DMLP * DENC);
  for (int i = tid; i < NAG * DENC; i += BDIM) { H[i] = 0.0f; C[i] = 0.0f; XE[i] = 0.0f; }
  __syncthreads();

  // ====================== ENCODER ======================
  for (int t = 0; t < TOBS; ++t) {
    // prefetch this step's spatial feature rows (HBM stream, hide behind GEMMs)
    for (int i = tid; i < NAG * 3 * 2; i += BDIM) {
      int mat = i / (NAG * 2);
      int row = (i >> 1) & (NAG - 1);
      int half = i & 1;
      const float* base = (mat == 0) ? p.dmat : (mat == 1) ? p.bmat : p.hmat;
      const float* addr = base + ((((size_t)b * NAG + row) * TOBS + t) * NAG) + half * 32;
      __builtin_prefetch(addr, 0, 3);
    }
    if (tid < NAG) MK[tid] = p.imask[(b * NAG + tid) * TOBS + t];
    // embedding (K=2, scalar) -> XE cols 0..15 (cols 16..31 remain zero)
    for (int i = tid; i < NAG * DEMB; i += BDIM) {
      int r = i >> 4, e = i & 15;
      const float* xr = p.x + (((size_t)b * NAG + r) * TOBS + t) * 2;
      XE[r * 32 + e] = p.emb_W[e * 2] * xr[0] + p.emb_W[e * 2 + 1] * xr[1] + p.emb_b[e];
    }
    __syncthreads();
    gemm_lstm(XE, WH + WO_ENCL_WIH, H, WH + WO_ENCL_WHH, G);
    __syncthreads();
    for (int i = tid; i < NAG * DENC; i += BDIM) {
      int r = i >> 5, c = i & 31;
      float gi = G[r * 128 + c]      + p.encl_bih[c]      + p.encl_bhh[c];
      float gf = G[r * 128 + 32 + c] + p.encl_bih[32 + c] + p.encl_bhh[32 + c];
      float gg = G[r * 128 + 64 + c] + p.encl_bih[64 + c] + p.encl_bhh[64 + c];
      float go = G[r * 128 + 96 + c] + p.encl_bih[96 + c] + p.encl_bhh[96 + c];
      float cn = sigm(gf) * C[i] + sigm(gi) * tanhf(gg);
      C[i] = cn;
      H[i] = sigm(go) * tanhf(cn);
    }
    __syncthreads();
    gemm_w<64, 32, 32, ACT_NONE>(H, DENC, WH + WO_ENC2ATT, 32, p.enc2att_b, HA, 32);
    __syncthreads();
    spatial_weights(WM, RS, MK, p.domain, true, p, b, t, CUR, PRV, HEAD);
    gemm_a<64, 32, 64>(WM, NAG, HA, 32, CAT, 64);     // wh -> cat[:,0:32]
    for (int i = tid; i < NAG * 32; i += BDIM) {      // h  -> cat[:,32:64]
      int r = i >> 5, d = i & 31;
      CAT[r * 64 + 32 + d] = HA[i];
    }
    __syncthreads();
    gemm_w<64, 32, 64, ACT_NONE>(CAT, 64, WH + WO_ENCSP, 64, p.encsp_b, HA, 32);
    __syncthreads();
    gemm_w<64, 32, 32, ACT_NONE>(HA, 32, WH + WO_ATT2ENC, 32, p.att2enc_b, H, DENC);
    __syncthreads();
    for (int i = tid; i < NAG * DENC; i += BDIM) {
      int r = i >> 5, d = i & 31;
      SEQ[r * (TOBS * DENC) + t * DENC + d] = H[i];
    }
    // output MLP: 32 -> 128 (WMMA, relu) -> 2 (scalar, tanh)
    gemm_w<64, 128, 32, ACT_RELU>(H, DENC, WH + WO_OUT1, DENC, p.out1_b, G, DMLP);
    __syncthreads();
    if (tid < NAG * 2) {
      int r = tid >> 1, k = tid & 1;
      float a = p.out2_b[k];
      for (int m = 0; m < DMLP; ++m) a += G[r * DMLP + m] * p.out2_W[k * DMLP + m];
      out[PRED_SZ + (((size_t)b * NAG + r) * TOBS + t) * 2 + k] = tanhf(a);
    }
    __syncthreads();
  }

  // ====================== DECODER INIT ======================
  if (tid < NAG * 2) {
    int r = tid >> 1, k = tid & 1;
    float xv = p.x[(((size_t)b * NAG + r) * TOBS + (TOBS - 1)) * 2 + k];
    XY[tid] = xv;  // xcur (normalized)
    float mk = p.imask[(b * NAG + r) * TOBS + (TOBS - 1)];
    CUR[tid] = (xv * p.var[b * 2 + k] + p.mean[b * 2 + k]) * mk;  // x_prev_init
  }
  for (int i = tid; i < NAG * DENC; i += BDIM) C[i] = 0.0f;  // h carried, c reset
  __syncthreads();

  // ====================== DECODER ======================
  for (int j = 0; j < TPRED; ++j) {
    if (tid < NAG) MK[tid] = p.omask[(b * NAG + tid) * TPRED + j];
    for (int i = tid; i < NAG * DEMB; i += BDIM) {
      int r = i >> 4, e = i & 15;
      XE[r * 32 + e] = p.emb_W[e * 2] * XY[r * 2] + p.emb_W[e * 2 + 1] * XY[r * 2 + 1] + p.emb_b[e];
    }
    __syncthreads();
    gemm_lstm(XE, WH + WO_DECL_WIH, H, WH + WO_DECL_WHH, G);
    __syncthreads();
    for (int i = tid; i < NAG * DENC; i += BDIM) {
      int r = i >> 5, c = i & 31;
      float gi = G[r * 128 + c]      + p.decl_bih[c]      + p.decl_bhh[c];
      float gf = G[r * 128 + 32 + c] + p.decl_bih[32 + c] + p.decl_bhh[32 + c];
      float gg = G[r * 128 + 64 + c] + p.decl_bih[64 + c] + p.decl_bhh[64 + c];
      float go = G[r * 128 + 96 + c] + p.decl_bih[96 + c] + p.decl_bhh[96 + c];
      float cn = sigm(gf) * C[i] + sigm(gi) * tanhf(gg);
      C[i] = cn;
      H[i] = sigm(go) * tanhf(cn);
    }
    __syncthreads();
    gemm_w<64, 32, 32, ACT_NONE>(H, DENC, WH + WO_DEC2ATT, 32, p.dec2att_b, HA, 32);
    __syncthreads();
    spatial_weights(WM, RS, MK, p.domain, (j == 0), p, b, TOBS - 1, CUR, PRV, HEAD);
    gemm_a<64, 32, 64>(WM, NAG, HA, 32, CAT, 64);
    for (int i = tid; i < NAG * 32; i += BDIM) {
      int r = i >> 5, d = i & 31;
      CAT[r * 64 + 32 + d] = HA[i];
    }
    __syncthreads();
    gemm_w<64, 32, 64, ACT_NONE>(CAT, 64, WH + WO_DECSP, 64, p.decsp_b, HA, 32);
    __syncthreads();
    gemm_w<64, 32, 32, ACT_NONE>(HA, 32, WH + WO_ATT2DEC, 32, p.att2dec_b, H, DENC);
    __syncthreads();
    // temporal attention over 8 encoder states (per track, scalar)
    if (tid < NAG) {
      float sc[TOBS];
      float mx = -1e30f;
      for (int o = 0; o < TOBS; ++o) {
        float s = 0.0f;
        for (int d = 0; d < DENC; ++d)
          s += H[tid * DENC + d] * SEQ[tid * (TOBS * DENC) + o * DENC + d];
        float im = p.imask[(b * NAG + tid) * TOBS + o];
        if (!(im > 0.0f)) s = -1e9f;
        sc[o] = s;
        mx = fmaxf(mx, s);
      }
      float ssum = 0.0f;
      for (int o = 0; o < TOBS; ++o) { float e = expf(sc[o] - mx); sc[o] = e; ssum += e; }
      float inv = 1.0f / ssum;
      for (int d = 0; d < DENC; ++d) {
        float a = 0.0f;
        for (int o = 0; o < TOBS; ++o) a += sc[o] * SEQ[tid * (TOBS * DENC) + o * DENC + d];
        CAT[tid * 64 + d] = a * inv;            // ctx
        CAT[tid * 64 + 32 + d] = H[tid * DENC + d];
      }
    }
    __syncthreads();
    gemm_w<64, 32, 64, ACT_TANH>(CAT, 64, WH + WO_TATT, 64, p.tatt_b, H, DENC);
    __syncthreads();
    gemm_w<64, 128, 32, ACT_RELU>(H, DENC, WH + WO_OUT1, DENC, p.out1_b, G, DMLP);
    __syncthreads();
    if (tid < NAG * 2) {
      int r = tid >> 1, k = tid & 1;
      float a = p.out2_b[k];
      for (int m = 0; m < DMLP; ++m) a += G[r * DMLP + m] * p.out2_W[k * DMLP + m];
      a = tanhf(a);
      out[(((size_t)b * NAG + r) * TPRED + j) * 2 + k] = a;
      XY[tid] = a;  // next xcur
      float nc = (a * p.var[b * 2 + k] + p.mean[b * 2 + k]) * MK[r];
      PRV[tid] = CUR[tid];  // x_prev <- old x_out_
      CUR[tid] = nc;        // cur   <- new x_out_
    }
    __syncthreads();
  }
}

// ---------------------------------------------------------------------------
extern "C" void kernel_launch(void* const* d_in, const int* in_sizes, int n_in,
                              void* d_out, int out_size, void* d_ws, size_t ws_size,
                              hipStream_t stream) {
  (void)in_sizes; (void)n_in; (void)d_ws; (void)ws_size; (void)out_size;
  P p;
  p.x         = (const float*)d_in[0];
  p.dmat      = (const float*)d_in[1];
  p.bmat      = (const float*)d_in[2];
  p.hmat      = (const float*)d_in[3];
  p.imask     = (const float*)d_in[4];
  p.omask     = (const float*)d_in[5];
  p.mean      = (const float*)d_in[6];
  p.var       = (const float*)d_in[7];
  p.emb_W     = (const float*)d_in[8];
  p.emb_b     = (const float*)d_in[9];
  p.encl_Wih  = (const float*)d_in[10];
  p.encl_Whh  = (const float*)d_in[11];
  p.encl_bih  = (const float*)d_in[12];
  p.encl_bhh  = (const float*)d_in[13];
  p.decl_Wih  = (const float*)d_in[14];
  p.decl_Whh  = (const float*)d_in[15];
  p.decl_bih  = (const float*)d_in[16];
  p.decl_bhh  = (const float*)d_in[17];
  p.enc2att_W = (const float*)d_in[18];
  p.enc2att_b = (const float*)d_in[19];
  p.att2enc_W = (const float*)d_in[20];
  p.att2enc_b = (const float*)d_in[21];
  p.dec2att_W = (const float*)d_in[22];
  p.dec2att_b = (const float*)d_in[23];
  p.att2dec_W = (const float*)d_in[24];
  p.att2dec_b = (const float*)d_in[25];
  p.encsp_W   = (const float*)d_in[26];
  p.encsp_b   = (const float*)d_in[27];
  p.decsp_W   = (const float*)d_in[28];
  p.decsp_b   = (const float*)d_in[29];
  p.tatt_W    = (const float*)d_in[30];
  p.tatt_b    = (const float*)d_in[31];
  p.out1_W    = (const float*)d_in[32];
  p.out1_b    = (const float*)d_in[33];
  p.out2_W    = (const float*)d_in[34];
  p.out2_b    = (const float*)d_in[35];
  p.domain    = (const float*)d_in[36];

  size_t shbytes = (size_t)SMEM_FLOATS * sizeof(float);
  traj_forward_kernel<<<BATCH, BDIM, shbytes, stream>>>(p, (float*)d_out);
}
